// GraphSAGENet_52613349376018
// MI455X (gfx1250) — matrix operand order repaired and checked by reading, not compiled, not run on hardware
//
#include <hip/hip_runtime.h>
#include <math.h>

// ---------------- problem constants (match reference) ----------------
#define BGR 32          // graphs
#define NPG 512         // nodes per graph
#define NTOT (BGR*NPG)  // 16384 nodes
#define EPG 4096
#define ETOT (BGR*EPG)  // 131072 edges
#define HH 4            // heads
#define INF_ 512
#define HID 128
#define OUTC 64
#define HHID (HH*HID)   // 512
#define KP1 308
#define KP2 185
#define SLOPE 0.2f

// ---------------- vector types ---------------------------------------
typedef __attribute__((ext_vector_type(16))) _Float16 v16h;
typedef __attribute__((ext_vector_type(8)))  _Float16 h8v;
typedef __attribute__((ext_vector_type(4)))  _Float16 h4v;
typedef __attribute__((ext_vector_type(8)))  float    v8f;

// ---------------- fp32 -> f16 conversion (vectorized) ----------------
__global__ __launch_bounds__(256)
void cvt_f16_k(const float* __restrict__ in, _Float16* __restrict__ out, long n4)
{
    long i = (long)blockIdx.x * 256 + threadIdx.x;
    if (i >= n4) return;
    float4 f = ((const float4*)in)[i];
    h4v h = { (_Float16)f.x, (_Float16)f.y, (_Float16)f.z, (_Float16)f.w };
    ((h4v*)out)[i] = h;
}

// W[K][Nt] fp32 -> Wt[Nt][K] f16 (coalesced writes along K)
__global__ __launch_bounds__(256)
void cvt_wt_k(const float* __restrict__ W, _Float16* __restrict__ Wt, int K, int Nt)
{
    int i = blockIdx.x * 256 + threadIdx.x;
    if (i >= K * Nt) return;
    int n = i / K, k = i - n * K;
    Wt[i] = (_Float16)W[(size_t)k * Nt + n];
}

// ---------------- WMMA GEMM (no LDS): C = A(f16) * Wt(f16)^T + bias --
// A: [M x K] row-major f16.  Wt: [Nt x K] f16 (i.e. W transposed).
// Fragments are loaded straight from global in a contiguous k order shared by A and B:
//   A frag: lane m=L%16, half=L/16; K = half*16 + e (two adjacent 8-half loads).
//   B frag: lane n=L%16,  K = half*16 + e -> one contiguous 16-half run.
// Requirements (all call sites): M%128==0, K%32==0, Nt%128==0.
#define BM 128
#define BN 128

__global__ __launch_bounds__(256)
void wmma_gemm_f16(const _Float16* __restrict__ A, const _Float16* __restrict__ Wt,
                   const float* __restrict__ bias, float* __restrict__ C,
                   int K, int Nt)
{
    const int tid  = threadIdx.x;
    const int lane = tid & 31;
    const int wv   = tid >> 5;
    const int wvm  = wv & 3;       // 4 waves along M, 32 rows each
    const int wvn  = wv >> 2;      // 2 waves along N, 64 cols each
    const int bm   = blockIdx.y * BM;
    const int bn   = blockIdx.x * BN;
    const int mrow = lane & 15;
    const int half = lane >> 4;

    v8f acc[2][4];
    v8f zero = {0.f,0.f,0.f,0.f,0.f,0.f,0.f,0.f};
    #pragma unroll
    for (int u = 0; u < 2; ++u)
        #pragma unroll
        for (int t = 0; t < 4; ++t) acc[u][t] = zero;

    const _Float16* a0 = A + (size_t)(bm + wvm * 32 + mrow) * K + half * 16;
    const _Float16* a1 = a0 + (size_t)16 * K;
    const _Float16* b0 = Wt + (size_t)(bn + wvn * 64 + mrow) * K + half * 16;

    for (int k0 = 0; k0 < K; k0 += 32) {
        if (k0 + 32 < K) {   // prefetch next k-step A rows -> global_prefetch_b8
            __builtin_prefetch(a0 + k0 + 32, 0, 1);
            __builtin_prefetch(a1 + k0 + 32, 0, 1);
        }
        v16h a[2];
        {
            h8v lo = *(const h8v*)(a0 + k0);
            h8v hi = *(const h8v*)(a0 + k0 + 8);
            a[0] = __builtin_shufflevector(lo, hi, 0,1,2,3,4,5,6,7,8,9,10,11,12,13,14,15);
        }
        {
            h8v lo = *(const h8v*)(a1 + k0);
            h8v hi = *(const h8v*)(a1 + k0 + 8);
            a[1] = __builtin_shufflevector(lo, hi, 0,1,2,3,4,5,6,7,8,9,10,11,12,13,14,15);
        }
        #pragma unroll
        for (int t = 0; t < 4; ++t) {
            // 16 contiguous halfs, 32B-aligned (K%32==0, row stride K*2B)
            v16h b = *(const v16h*)(b0 + (size_t)t * 16 * K + k0);
            acc[0][t] = __builtin_amdgcn_wmma_f32_16x16x32_f16(
                false, a[0], false, b, (short)0, acc[0][t], false, false);
            acc[1][t] = __builtin_amdgcn_wmma_f32_16x16x32_f16(
                false, a[1], false, b, (short)0, acc[1][t], false, false);
        }
    }

    // C/D layout: lane L, vgpr i -> row = (L/16)*8 + i, col = L%16
    #pragma unroll
    for (int t = 0; t < 4; ++t) {
        int col = bn + wvn * 64 + t * 16 + mrow;
        float bv = bias[col];
        #pragma unroll
        for (int u = 0; u < 2; ++u) {
            #pragma unroll
            for (int i = 0; i < 8; ++i) {
                int row = bm + wvm * 32 + u * 16 + half * 8 + i;
                C[(size_t)row * Nt + col] = acc[u][t][i] + bv;
            }
        }
    }
}

// ---------------- attention helpers ----------------------------------
__device__ __forceinline__ unsigned f32_ord(float f) {
    unsigned u = __float_as_uint(f);
    return (u >> 31) ? ~u : (u | 0x80000000u);
}
__device__ __forceinline__ float ord_f32(unsigned u) {
    return (u >> 31) ? __uint_as_float(u & 0x7FFFFFFFu) : __uint_as_float(~u);
}
__device__ __forceinline__ void edge_sd(int e, int Eo, const int* src, const int* dst,
                                        const int* valid, int& s, int& d, bool& ok) {
    if (e < Eo) {
        ok = (valid == nullptr) || (valid[e] != 0);
        s = src[e]; d = dst[e];
    } else { ok = true; s = d = e - Eo; }  // self loops, always valid
}

// per-(edge,head) logits + segment max; one wave per (e,h)
__global__ __launch_bounds__(256)
void edge_logit_k(const int* __restrict__ src, const int* __restrict__ dst,
                  const int* __restrict__ valid,
                  const float* __restrict__ xl, const float* __restrict__ xr,
                  const float* __restrict__ att,
                  float* __restrict__ logit, unsigned* __restrict__ segmax,
                  int Eo, int n)
{
    int w = (blockIdx.x * 256 + threadIdx.x) >> 5;
    int lane = threadIdx.x & 31;
    if (w >= (Eo + n) * HH) return;
    int e = w >> 2, h = w & 3;
    int s, d; bool ok;
    edge_sd(e, Eo, src, dst, valid, s, d, ok);
    if (!ok) return;
    const float* pl = xl + (size_t)s * HHID + h * HID;
    const float* pr = xr + (size_t)d * HHID + h * HID;
    const float* pa = att + h * HID;
    float acc = 0.f;
    #pragma unroll
    for (int j = 0; j < 4; ++j) {
        int c = lane + j * 32;
        float v = pl[c] + pr[c];
        float lr = v > 0.f ? v : SLOPE * v;
        acc += lr * pa[c];
    }
    for (int off = 16; off > 0; off >>= 1) acc += __shfl_xor(acc, off);
    if (lane == 0) {
        logit[w] = acc;
        atomicMax(&segmax[d * HH + h], f32_ord(acc));
    }
}

// exp(logit - segmax[dst]) + segment sum; one thread per (e,h)
__global__ __launch_bounds__(256)
void edge_exp_k(const int* __restrict__ src, const int* __restrict__ dst,
                const int* __restrict__ valid,
                const float* __restrict__ logit, const unsigned* __restrict__ segmax,
                float* __restrict__ alpha, float* __restrict__ segsum, int Eo, int n)
{
    int w = blockIdx.x * 256 + threadIdx.x;
    if (w >= (Eo + n) * HH) return;
    int e = w >> 2, h = w & 3;
    int s, d; bool ok;
    edge_sd(e, Eo, src, dst, valid, s, d, ok);
    if (!ok) return;
    float ex = __expf(logit[w] - ord_f32(segmax[d * HH + h]));
    alpha[w] = ex;
    atomicAdd(&segsum[d * HH + h], ex);
}

// normalized message scatter: accum[dst] += alpha * xl[src]; wave per (e,h)
__global__ __launch_bounds__(256)
void edge_msg_k(const int* __restrict__ src, const int* __restrict__ dst,
                const int* __restrict__ valid,
                const float* __restrict__ xl, const float* __restrict__ alpha,
                const float* __restrict__ segsum, float* __restrict__ accum,
                int Eo, int n)
{
    int w = (blockIdx.x * 256 + threadIdx.x) >> 5;
    int lane = threadIdx.x & 31;
    if (w >= (Eo + n) * HH) return;
    int e = w >> 2, h = w & 3;
    int s, d; bool ok;
    edge_sd(e, Eo, src, dst, valid, s, d, ok);
    if (!ok) return;
    float a = alpha[w] / (segsum[d * HH + h] + 1e-16f);
    const float* px = xl + (size_t)s * HHID + h * HID;
    float* po = accum + (size_t)d * HHID + h * HID;
    #pragma unroll
    for (int j = 0; j < 4; ++j) {
        int c = lane + j * 32;
        atomicAdd(&po[c], a * px[c]);
    }
}

// head mean + bias + ReLU
__global__ __launch_bounds__(256)
void head_mean_k(const float* __restrict__ accum, const float* __restrict__ bias,
                 float* __restrict__ xout, int n)
{
    int i = blockIdx.x * 256 + threadIdx.x;
    if (i >= n * HID) return;
    int node = i >> 7, c = i & 127;
    const float* p = accum + (size_t)node * HHID + c;
    float v = (p[0] + p[HID] + p[2*HID] + p[3*HID]) * 0.25f + bias[c];
    xout[i] = v > 0.f ? v : 0.f;
}

// ---------------- SAGE score -----------------------------------------
__global__ __launch_bounds__(256)
void sage_agg_k(const int* __restrict__ src, const int* __restrict__ dst,
                const int* __restrict__ valid, const float* __restrict__ x,
                float* __restrict__ sums, float* __restrict__ cnt, int Eo)
{
    int e = (blockIdx.x * 256 + threadIdx.x) >> 5;
    int lane = threadIdx.x & 31;
    if (e >= Eo) return;
    if (valid && !valid[e]) return;
    int s = src[e], d = dst[e];
    const float* px = x + (size_t)s * HID;
    float* po = sums + (size_t)d * HID;
    #pragma unroll
    for (int j = 0; j < 4; ++j) { int c = lane + j * 32; atomicAdd(&po[c], px[c]); }
    if (lane == 0) atomicAdd(&cnt[d], 1.0f);
}

__global__ __launch_bounds__(256)
void sage_score_k(const float* __restrict__ x, const float* __restrict__ sums,
                  const float* __restrict__ cnt, const float* __restrict__ Wn,
                  const float* __restrict__ Wr, const float* __restrict__ bs,
                  float* __restrict__ score, int n)
{
    int nd = (blockIdx.x * 256 + threadIdx.x) >> 5;
    int lane = threadIdx.x & 31;
    if (nd >= n) return;
    float cc = fmaxf(cnt[nd], 1.0f);
    float acc = 0.f;
    #pragma unroll
    for (int j = 0; j < 4; ++j) {
        int c = lane + j * 32;
        float mean = sums[(size_t)nd * HID + c] / cc;
        acc += mean * Wn[c] + x[(size_t)nd * HID + c] * Wr[c];
    }
    for (int off = 16; off > 0; off >>= 1) acc += __shfl_xor(acc, off);
    if (lane == 0) score[nd] = acc + bs[0];
}

// ---------------- SAGPooling top-k (rank select, order-invariant) -----
__global__ __launch_bounds__(512)
void pool_k(const float* __restrict__ score, int* __restrict__ nmap,
            int* __restrict__ old_of_new, float* __restrict__ gate,
            int npg, int k)
{
    __shared__ float sc[512];
    __shared__ int kp[512];
    int g = blockIdx.x, t = threadIdx.x;
    if (t < npg) sc[t] = score[g * npg + t];
    __syncthreads();
    if (t < npg) {
        float my = sc[t]; int rank = 0;
        for (int j = 0; j < npg; ++j) {
            float o = sc[j];
            rank += (o > my) || (o == my && j < t);
        }
        kp[t] = (rank < k) ? 1 : 0;
    } else kp[t] = 0;
    __syncthreads();
    if (t < npg) {
        int pre = 0;
        for (int j = 0; j < t; ++j) pre += kp[j];
        int old = g * npg + t;
        if (kp[t]) {
            int nid = g * k + pre;
            nmap[old] = nid;
            old_of_new[nid] = old;
            gate[nid] = tanhf(sc[t]);
        } else nmap[old] = -1;
    }
}

__global__ __launch_bounds__(256)
void gather_k(const float* __restrict__ xold, const int* __restrict__ old_of_new,
              const float* __restrict__ gate, float* __restrict__ xnew, int nnew)
{
    int i = blockIdx.x * 256 + threadIdx.x;
    if (i >= nnew * HID) return;
    int node = i >> 7, c = i & 127;
    xnew[i] = xold[(size_t)old_of_new[node] * HID + c] * gate[node];
}

__global__ __launch_bounds__(256)
void remap_k(const int* __restrict__ sin, const int* __restrict__ din,
             const int* __restrict__ vin, const int* __restrict__ nmap,
             int* __restrict__ sout, int* __restrict__ dout, int* __restrict__ vout,
             int Eo)
{
    int e = blockIdx.x * 256 + threadIdx.x;
    if (e >= Eo) return;
    bool ok = (vin == nullptr) || (vin[e] != 0);
    int ns = -1, nd = -1;
    if (ok) { ns = nmap[sin[e]]; nd = nmap[din[e]]; }
    ok = ok && ns >= 0 && nd >= 0;
    sout[e] = ok ? ns : 0;
    dout[e] = ok ? nd : 0;
    vout[e] = ok ? 1 : 0;
}

// ---------------- global max pool (contiguous equal groups) -----------
__global__ __launch_bounds__(128)
void gmp_k(const float* __restrict__ x, float* __restrict__ out, int per)
{
    int g = blockIdx.x, c = threadIdx.x;
    float m = -INFINITY;
    for (int i = 0; i < per; ++i)
        m = fmaxf(m, x[((size_t)(g * per + i)) * HID + c]);
    out[g * HID + c] = m;
}

// ---------------- final FC: relu((g1+g2+g3) @ Wf + bf) ----------------
__global__ __launch_bounds__(256)
void final_k(const float* __restrict__ g1, const float* __restrict__ g2,
             const float* __restrict__ g3, const float* __restrict__ Wf,
             const float* __restrict__ bf, float* __restrict__ out)
{
    int i = blockIdx.x * 256 + threadIdx.x;
    if (i >= BGR * OUTC) return;
    int b = i / OUTC, o = i % OUTC;
    float acc = bf[o];
    for (int c = 0; c < HID; ++c) {
        float xv = g1[b * HID + c] + g2[b * HID + c] + g3[b * HID + c];
        acc += xv * Wf[c * OUTC + o];
    }
    out[i] = acc > 0.f ? acc : 0.f;
}

// ---------------- orchestration ---------------------------------------
extern "C" void kernel_launch(void* const* d_in, const int* in_sizes, int n_in,
                              void* d_out, int out_size, void* d_ws, size_t ws_size,
                              hipStream_t stream)
{
    const float* x_feat = (const float*)d_in[0];
    const int*   eidx   = (const int*)d_in[1];
    const float* Wl1 = (const float*)d_in[3];  const float* bl1 = (const float*)d_in[4];
    const float* Wr1 = (const float*)d_in[5];  const float* br1 = (const float*)d_in[6];
    const float* att1 = (const float*)d_in[7]; const float* bias1 = (const float*)d_in[8];
    const float* Wl2 = (const float*)d_in[9];  const float* bl2 = (const float*)d_in[10];
    const float* Wr2 = (const float*)d_in[11]; const float* br2 = (const float*)d_in[12];
    const float* att2 = (const float*)d_in[13]; const float* bias2 = (const float*)d_in[14];
    const float* Wsn = (const float*)d_in[15]; const float* Wsr = (const float*)d_in[16];
    const float* bs  = (const float*)d_in[17];
    const float* Wf  = (const float*)d_in[18]; const float* bf = (const float*)d_in[19];

    const int* src = eidx;
    const int* dst = eidx + ETOT;

    // bump allocator over d_ws
    char* base = (char*)d_ws;
    size_t off = 0;
    auto alloc = [&](size_t bytes) -> void* {
        off = (off + 255) & ~(size_t)255;
        void* p = base + off;
        off += bytes;
        return p;
    };
    float*    xl     = (float*)   alloc((size_t)NTOT * HHID * 4);
    float*    xr     = (float*)   alloc((size_t)NTOT * HHID * 4);
    float*    accum  = (float*)   alloc((size_t)NTOT * HHID * 4);
    _Float16* xh     = (_Float16*)alloc((size_t)NTOT * INF_ * 2);   // f16 A operand
    _Float16* wlh    = (_Float16*)alloc((size_t)INF_ * HHID * 2);   // f16 Wl^T
    _Float16* wrh    = (_Float16*)alloc((size_t)INF_ * HHID * 2);   // f16 Wr^T
    unsigned* segmax = (unsigned*)alloc((size_t)NTOT * HH * 4);
    float*    segsum = (float*)   alloc((size_t)NTOT * HH * 4);
    float*    logit  = (float*)   alloc((size_t)(ETOT + NTOT) * HH * 4);
    float*    alpha  = (float*)   alloc((size_t)(ETOT + NTOT) * HH * 4);
    float*    xA     = (float*)   alloc((size_t)NTOT * HID * 4);
    float*    xB     = (float*)   alloc((size_t)NTOT * HID * 4);
    float*    sums   = (float*)   alloc((size_t)NTOT * HID * 4);
    float*    cnt    = (float*)   alloc((size_t)NTOT * 4);
    float*    score  = (float*)   alloc((size_t)NTOT * 4);
    int*      nmap   = (int*)     alloc((size_t)NTOT * 4);
    int*      oon    = (int*)     alloc((size_t)NTOT * 4);
    float*    gate   = (float*)   alloc((size_t)NTOT * 4);
    int*      s2     = (int*)     alloc((size_t)ETOT * 4);
    int*      d2     = (int*)     alloc((size_t)ETOT * 4);
    int*      v2     = (int*)     alloc((size_t)ETOT * 4);
    float*    g1     = (float*)   alloc((size_t)BGR * HID * 4);
    float*    g2     = (float*)   alloc((size_t)BGR * HID * 4);
    float*    g3     = (float*)   alloc((size_t)BGR * HID * 4);

    const int n2 = BGR * KP1;   // 9856 = 77*128
    const int n3 = BGR * KP2;   // 5920

    auto cdiv = [](long a, long b) { return (int)((a + b - 1) / b); };

    // ===== layer 1: GATv2 on full graph =====
    cvt_f16_k<<<cdiv((long)NTOT * INF_ / 4, 256), 256, 0, stream>>>(x_feat, xh, (long)NTOT * INF_ / 4);
    cvt_wt_k <<<cdiv((long)INF_ * HHID, 256), 256, 0, stream>>>(Wl1, wlh, INF_, HHID);
    cvt_wt_k <<<cdiv((long)INF_ * HHID, 256), 256, 0, stream>>>(Wr1, wrh, INF_, HHID);
    {
        dim3 g(HHID / BN, NTOT / BM);   // (4, 128)
        wmma_gemm_f16<<<g, 256, 0, stream>>>(xh, wlh, bl1, xl, INF_, HHID);
        wmma_gemm_f16<<<g, 256, 0, stream>>>(xh, wrh, br1, xr, INF_, HHID);
    }
    hipMemsetAsync(segmax, 0, (size_t)NTOT * HH * 4, stream);
    hipMemsetAsync(segsum, 0, (size_t)NTOT * HH * 4, stream);
    hipMemsetAsync(accum,  0, (size_t)NTOT * HHID * 4, stream);
    {
        long tot = (long)(ETOT + NTOT) * HH;
        edge_logit_k<<<cdiv(tot * 32, 256), 256, 0, stream>>>(src, dst, nullptr, xl, xr, att1, logit, segmax, ETOT, NTOT);
        edge_exp_k  <<<cdiv(tot, 256),      256, 0, stream>>>(src, dst, nullptr, logit, segmax, alpha, segsum, ETOT, NTOT);
        edge_msg_k  <<<cdiv(tot * 32, 256), 256, 0, stream>>>(src, dst, nullptr, xl, alpha, segsum, accum, ETOT, NTOT);
        head_mean_k <<<cdiv((long)NTOT * HID, 256), 256, 0, stream>>>(accum, bias1, xA, NTOT);
    }
    gmp_k<<<BGR, HID, 0, stream>>>(xA, g1, NPG);

    // SAGE score on full graph
    hipMemsetAsync(sums, 0, (size_t)NTOT * HID * 4, stream);
    hipMemsetAsync(cnt,  0, (size_t)NTOT * 4, stream);
    sage_agg_k  <<<cdiv((long)ETOT * 32, 256), 256, 0, stream>>>(src, dst, nullptr, xA, sums, cnt, ETOT);
    sage_score_k<<<cdiv((long)NTOT * 32, 256), 256, 0, stream>>>(xA, sums, cnt, Wsn, Wsr, bs, score, NTOT);

    // pool 1: N -> n2
    pool_k  <<<BGR, 512, 0, stream>>>(score, nmap, oon, gate, NPG, KP1);
    gather_k<<<cdiv((long)n2 * HID, 256), 256, 0, stream>>>(xA, oon, gate, xB, n2);
    remap_k <<<cdiv((long)ETOT, 256), 256, 0, stream>>>(src, dst, nullptr, nmap, s2, d2, v2, ETOT);
    gmp_k<<<BGR, HID, 0, stream>>>(xB, g2, KP1);

    // ===== layer 2: GATv2 on pooled graph =====
    cvt_f16_k<<<cdiv((long)n2 * HID / 4, 256), 256, 0, stream>>>(xB, xh, (long)n2 * HID / 4);
    cvt_wt_k <<<cdiv((long)HID * HHID, 256), 256, 0, stream>>>(Wl2, wlh, HID, HHID);
    cvt_wt_k <<<cdiv((long)HID * HHID, 256), 256, 0, stream>>>(Wr2, wrh, HID, HHID);
    {
        dim3 g(HHID / BN, n2 / BM);     // (4, 77)
        wmma_gemm_f16<<<g, 256, 0, stream>>>(xh, wlh, bl2, xl, HID, HHID);
        wmma_gemm_f16<<<g, 256, 0, stream>>>(xh, wrh, br2, xr, HID, HHID);
    }
    hipMemsetAsync(segmax, 0, (size_t)NTOT * HH * 4, stream);
    hipMemsetAsync(segsum, 0, (size_t)NTOT * HH * 4, stream);
    hipMemsetAsync(accum,  0, (size_t)NTOT * HHID * 4, stream);
    {
        long tot = (long)(ETOT + n2) * HH;
        edge_logit_k<<<cdiv(tot * 32, 256), 256, 0, stream>>>(s2, d2, v2, xl, xr, att2, logit, segmax, ETOT, n2);
        edge_exp_k  <<<cdiv(tot, 256),      256, 0, stream>>>(s2, d2, v2, logit, segmax, alpha, segsum, ETOT, n2);
        edge_msg_k  <<<cdiv(tot * 32, 256), 256, 0, stream>>>(s2, d2, v2, xl, alpha, segsum, accum, ETOT, n2);
        head_mean_k <<<cdiv((long)n2 * HID, 256), 256, 0, stream>>>(accum, bias2, xA, n2);
    }

    // SAGE score on pooled graph
    hipMemsetAsync(sums, 0, (size_t)NTOT * HID * 4, stream);
    hipMemsetAsync(cnt,  0, (size_t)NTOT * 4, stream);
    sage_agg_k  <<<cdiv((long)ETOT * 32, 256), 256, 0, stream>>>(s2, d2, v2, xA, sums, cnt, ETOT);
    sage_score_k<<<cdiv((long)n2 * 32, 256), 256, 0, stream>>>(xA, sums, cnt, Wsn, Wsr, bs, score, n2);

    // pool 2: n2 -> n3 (pooled edges unused downstream)
    pool_k  <<<BGR, 512, 0, stream>>>(score, nmap, oon, gate, KP1, KP2);
    gather_k<<<cdiv((long)n3 * HID, 256), 256, 0, stream>>>(xA, oon, gate, xB, n3);
    gmp_k<<<BGR, HID, 0, stream>>>(xB, g3, KP2);

    // final
    final_k<<<cdiv((long)BGR * OUTC, 256), 256, 0, stream>>>(g1, g2, g3, Wf, bf, (float*)d_out);
}
